// PyGRandLANet_33217277067473
// MI455X (gfx1250) — matrix operand order repaired and checked by reading, not compiled, not run on hardware
//
#include <hip/hip_runtime.h>
#include <cstdint>
#include <cstddef>

typedef __attribute__((ext_vector_type(16))) _Float16 v16h;
typedef __attribute__((ext_vector_type(8)))  _Float16 v8h;
typedef __attribute__((ext_vector_type(8)))  float    v8f;

#define LRELU_SLOPE 0.2f

__device__ __forceinline__ float leakyf(float v) { return v > 0.f ? v : LRELU_SLOPE * v; }

// ---------------------------------------------------------------------------
// f32 -> f16 cast
// ---------------------------------------------------------------------------
__global__ void k_cast_f16(const float* __restrict__ in, _Float16* __restrict__ out, int n) {
  int i = blockIdx.x * blockDim.x + threadIdx.x;
  if (i < n) out[i] = (_Float16)in[i];
}

// ---------------------------------------------------------------------------
// Pack W [Kd, Nc] (row-major f32) into WMMA B-fragment layout (f16):
//   Wp[((ct*nK + kk)*32 + lane)*16 + i] = W[kb + (lane>=16)*16 + i, ct*16 + lane%16]
// so the GEMM loads one contiguous v16h per lane per K-step.
// ---------------------------------------------------------------------------
__global__ void k_pack_w(const float* __restrict__ W, _Float16* __restrict__ Wp, int Kd, int Nc) {
  int nK = Kd >> 5;
  int total = Kd * Nc;
  int idx = blockIdx.x * blockDim.x + threadIdx.x;
  if (idx >= total) return;
  int i    = idx & 15;
  int lane = (idx >> 4) & 31;
  int t    = idx >> 9;          // = ct*nK + kk
  int kk   = t % nK;
  int ct   = t / nK;
  int col  = ct * 16 + (lane & 15);
  int k    = (kk << 5) + ((lane >> 4) << 4) + i;
  Wp[idx] = (_Float16)W[(size_t)k * Nc + col];
}

// ---------------------------------------------------------------------------
// WMMA GEMM: C[M,Nc] = A[M,Kd](f16) @ Wp(packed f16) (+ bias), f32 out.
// One wave computes a 16 x (16*NT) output tile: the A fragment (the E-sized,
// HBM-resident operand) is loaded ONCE and reused across NT B tiles (weights,
// L2-resident) -> A HBM traffic divided by NT vs a 16x16 tiling.
// Kd multiple of 32; M multiple of 16; Nc multiple of 16*NT (guaranteed).
// Early-out is wave-uniform so EXEC is all-ones at every WMMA (ISA 7.12).
// ---------------------------------------------------------------------------
template <int NT>
__global__ void k_wmma_gemm(const _Float16* __restrict__ A, const _Float16* __restrict__ Wp,
                            const float* __restrict__ bias, float* __restrict__ C,
                            int M, int Kd, int Nc) {
  int lane = threadIdx.x & 31;
  int wave = blockIdx.x * (blockDim.x >> 5) + (threadIdx.x >> 5);
  int nK = Kd >> 5;
  int groupsN = Nc / (16 * NT);
  int tileM = wave / groupsN;
  int tg    = wave % groupsN;
  if (tileM * 16 >= M) return;                  // wave-uniform
  int hsel = lane >> 4;                          // 0 or 1
  int row  = tileM * 16 + (lane & 15);
  const _Float16* arow = A + (size_t)row * Kd;
  const _Float16* wpg  = Wp + ((size_t)tg * NT * nK) * 512 + (size_t)lane * 16;
  v8f acc[NT];
#pragma unroll
  for (int t = 0; t < NT; ++t) acc[t] = (v8f){};
  for (int kk = 0; kk < nK; ++kk) {
    int kb = kk << 5;
    // A 16x32 f16 fragment (contiguous, matches packed B): lanes<16 hold K {0..15}, lanes>=16 hold {16..31}
    v8h alo = *(const v8h*)(arow + kb + hsel * 16);
    v8h ahi = *(const v8h*)(arow + kb + hsel * 16 + 8);
    v16h a = __builtin_shufflevector(alo, ahi, 0,1,2,3,4,5,6,7,8,9,10,11,12,13,14,15);
#pragma unroll
    for (int t = 0; t < NT; ++t) {
      v16h b = *(const v16h*)(wpg + ((size_t)t * nK + kk) * 512);
      acc[t] = __builtin_amdgcn_wmma_f32_16x16x32_f16(false, a, false, b, (short)0, acc[t], false, false);
    }
  }
  // C/D layout: VGPR v -> row tileM*16 + v + 8*hsel, col = tileN*16 + lane%16
#pragma unroll
  for (int t = 0; t < NT; ++t) {
    int col = (tg * NT + t) * 16 + (lane & 15);
    float bv = bias ? bias[col] : 0.f;
#pragma unroll
    for (int v = 0; v < 8; ++v) {
      int r = tileM * 16 + v + hsel * 8;
      C[(size_t)r * Nc + col] = acc[t][v] + bv;
    }
  }
}

// ---------------------------------------------------------------------------
// Per-column batch stats (training-mode BN): mu[c], rsqrt(var+eps)[c].
// One block per column; deterministic strided + LDS tree reduction.
// ---------------------------------------------------------------------------
__global__ void k_colstats(const float* __restrict__ C, float* __restrict__ mu,
                           float* __restrict__ rv, int M, int Nc) {
  __shared__ float ss[256], sq[256];
  int c = blockIdx.x;
  int t = threadIdx.x;
  float s = 0.f, s2 = 0.f;
  for (int r = t; r < M; r += 256) {
    float v = C[(size_t)r * Nc + c];
    s += v; s2 += v * v;
  }
  ss[t] = s; sq[t] = s2;
  __syncthreads();
  for (int o = 128; o > 0; o >>= 1) {
    if (t < o) { ss[t] += ss[t + o]; sq[t] += sq[t + o]; }
    __syncthreads();
  }
  if (t == 0) {
    float m = ss[0] / (float)M;
    float var = sq[0] / (float)M - m * m;
    mu[c] = m;
    rv[c] = rsqrtf(var + 1e-6f);
  }
}

// ---------------------------------------------------------------------------
// Fused BN (+ optional LeakyReLU) with optional f32 and/or f16 outputs.
// ---------------------------------------------------------------------------
__global__ void k_bn(const float* __restrict__ C, const float* __restrict__ mu,
                     const float* __restrict__ rv, const float* __restrict__ g,
                     const float* __restrict__ be, float* __restrict__ outF,
                     _Float16* __restrict__ outH, int total, int Nc, int act) {
  int idx = blockIdx.x * blockDim.x + threadIdx.x;
  if (idx >= total) return;
  int c = idx % Nc;
  float v = (C[idx] - mu[c]) * rv[c] * g[c] + be[c];
  if (act) v = leakyf(v);
  if (outF) outF[idx] = v;
  if (outH) outH[idx] = (_Float16)v;
}

// ---------------------------------------------------------------------------
// Darboux frame features per edge -> rel[E,8]
// ---------------------------------------------------------------------------
__device__ __forceinline__ void cross3(const float a[3], const float b[3], float o[3]) {
  o[0] = a[1]*b[2] - a[2]*b[1];
  o[1] = a[2]*b[0] - a[0]*b[2];
  o[2] = a[0]*b[1] - a[1]*b[0];
}
__device__ __forceinline__ float dot3(const float a[3], const float b[3]) {
  return a[0]*b[0] + a[1]*b[1] + a[2]*b[2];
}
__device__ __forceinline__ float norm3(const float a[3]) { return sqrtf(dot3(a, a)); }

__global__ void k_darboux(const float* __restrict__ pos, const float* __restrict__ nrm,
                          const int* __restrict__ ei, float* __restrict__ rel, int Ec) {
  int e = blockIdx.x * blockDim.x + threadIdx.x;
  if (e >= Ec) return;
  int s = ei[e], d = ei[Ec + e];
  float pi[3], pj[3], ni[3], nj[3], pd[3];
#pragma unroll
  for (int i = 0; i < 3; ++i) {
    pi[i] = pos[3 * d + i];  pj[i] = pos[3 * s + i];
    ni[i] = nrm[3 * d + i];  nj[i] = nrm[3 * s + i];
    pd[i] = pj[i] - pi[i];
  }
  float dist = norm3(pd), lni = norm3(ni), lnj = norm3(nj);
  float uq[3], vq[3], uk[3], vk[3];
  cross3(pd, ni, uq); cross3(uq, ni, vq);
  cross3(pd, nj, uk); cross3(uk, nj, vk);
  float luq = norm3(uq), lvq = norm3(vq), luk = norm3(uk), lvk = norm3(vk);
  float* r = rel + (size_t)e * 8;
  r[0] = dist;
  r[1] = dot3(pd, ni) / (dist * lni + 1e-10f);
  r[2] = dot3(pd, nj) / (dist * lnj + 1e-10f);
  r[3] = dot3(ni, nj) / (lni * lnj + 1e-10f);
  r[4] = dot3(uq, uk) / (luq * luk + 1e-10f);
  r[5] = dot3(vq, vk) / (lvq * lvk + 1e-10f);
  r[6] = dot3(uq, vk) / (luq * lvk + 1e-10f);
  r[7] = dot3(vq, uk) / (lvq * luk + 1e-10f);
}

// ---------------------------------------------------------------------------
// Encoder GEMM, K=8 (too small for WMMA): out[E,Nc] = rel[E,8] @ W[8,Nc] + b
// ---------------------------------------------------------------------------
__global__ void k_enc(const float* __restrict__ rel, const float* __restrict__ W,
                      const float* __restrict__ bias, float* __restrict__ out,
                      int Ec, int Nc) {
  int idx = blockIdx.x * blockDim.x + threadIdx.x;
  if (idx >= Ec * Nc) return;
  int e = idx / Nc, c = idx % Nc;
  const float* r = rel + (size_t)e * 8;
  float acc = bias[c];
#pragma unroll
  for (int i = 0; i < 8; ++i) acc += r[i] * W[i * Nc + c];
  out[idx] = acc;
}

// ---------------------------------------------------------------------------
// local = concat(h[src], lse) cast to f16 for the attention WMMA GEMM
// ---------------------------------------------------------------------------
__global__ void k_local(const float* __restrict__ hF, const float* __restrict__ lse,
                        const int* __restrict__ ei, _Float16* __restrict__ localH,
                        int Ec, int ch) {
  int c2 = ch * 2;
  int idx = blockIdx.x * blockDim.x + threadIdx.x;
  if (idx >= Ec * c2) return;
  int e = idx / c2, j = idx % c2;
  float v = (j < ch) ? hF[(size_t)ei[e] * ch + j] : lse[(size_t)e * ch + (j - ch)];
  localH[idx] = (_Float16)v;
}

// ---------------------------------------------------------------------------
// Segment softmax + weighted aggregation. Exploits dst = repeat(arange(N), K):
// node n's edges are rows [n*K, n*K+K). local recomputed (gather + lse) to
// avoid a 128 MB f32 buffer. Deterministic (no atomics).
// out[n,j] = sum_k exp(att[e_k,j]-max)*local[e_k,j] / (sum_k exp(...) + 1e-16)
// ---------------------------------------------------------------------------
__global__ void k_agg(const float* __restrict__ att, const float* __restrict__ hF,
                      const float* __restrict__ lse, const int* __restrict__ ei,
                      _Float16* __restrict__ aggH, int Nn, int Kn, int c, int ch) {
  int idx = blockIdx.x * blockDim.x + threadIdx.x;
  if (idx >= Nn * c) return;
  int n = idx / c, j = idx % c;
  size_t ebase = (size_t)n * Kn;
  float mx = -3.0e38f;
  for (int k = 0; k < Kn; ++k) mx = fmaxf(mx, att[(ebase + k) * c + j]);
  float s = 0.f, acc = 0.f;
  for (int k = 0; k < Kn; ++k) {
    size_t e = ebase + k;
    float ev = __expf(att[e * c + j] - mx);
    float lv = (j < ch) ? hF[(size_t)ei[e] * ch + j] : lse[e * ch + (j - ch)];
    s += ev;
    acc += ev * lv;
  }
  aggH[idx] = (_Float16)(acc / (s + 1e-16f));
}

// ---------------------------------------------------------------------------
// Final: BN(mlp2) + BN(shortcut), add, LeakyReLU -> d_out (f32)
// ---------------------------------------------------------------------------
__global__ void k_final(const float* __restrict__ h2raw, const float* __restrict__ scraw,
                        const float* __restrict__ mu2, const float* __restrict__ rv2,
                        const float* __restrict__ g2, const float* __restrict__ be2,
                        const float* __restrict__ mus, const float* __restrict__ rvs,
                        const float* __restrict__ gs, const float* __restrict__ bes,
                        float* __restrict__ out, int total, int Nc) {
  int idx = blockIdx.x * blockDim.x + threadIdx.x;
  if (idx >= total) return;
  int c = idx % Nc;
  float a = (h2raw[idx] - mu2[c]) * rv2[c] * g2[c] + be2[c];
  float b = (scraw[idx] - mus[c]) * rvs[c] * gs[c] + bes[c];
  out[idx] = leakyf(a + b);
}

// ===========================================================================
extern "C" void kernel_launch(void* const* d_in, const int* in_sizes, int n_in,
                              void* d_out, int out_size, void* d_ws, size_t ws_size,
                              hipStream_t stream) {
  (void)n_in; (void)out_size; (void)ws_size;
  const float* x        = (const float*)d_in[0];
  const float* pos      = (const float*)d_in[1];
  const float* normals  = (const float*)d_in[2];
  const int*   ei       = (const int*)d_in[3];
  const float* mlp1_w   = (const float*)d_in[4];
  const float* mlp1_b   = (const float*)d_in[5];
  const float* mlp1_g   = (const float*)d_in[6];
  const float* mlp1_be  = (const float*)d_in[7];
  const float* sc_w     = (const float*)d_in[8];
  const float* sc_b     = (const float*)d_in[9];
  const float* sc_g     = (const float*)d_in[10];
  const float* sc_be    = (const float*)d_in[11];
  const float* mlp2_w   = (const float*)d_in[12];
  const float* mlp2_b   = (const float*)d_in[13];
  const float* mlp2_g   = (const float*)d_in[14];
  const float* mlp2_be  = (const float*)d_in[15];
  const float* e1_w     = (const float*)d_in[16];
  const float* e1_b     = (const float*)d_in[17];
  const float* e1_g     = (const float*)d_in[18];
  const float* e1_be    = (const float*)d_in[19];
  const float* att1_w   = (const float*)d_in[20];
  const float* p1_w     = (const float*)d_in[21];
  const float* p1_b     = (const float*)d_in[22];
  const float* p1_g     = (const float*)d_in[23];
  const float* p1_be    = (const float*)d_in[24];
  const float* e2_w     = (const float*)d_in[25];
  const float* e2_b     = (const float*)d_in[26];
  const float* e2_g     = (const float*)d_in[27];
  const float* e2_be    = (const float*)d_in[28];
  const float* att2_w   = (const float*)d_in[29];
  const float* p2_w     = (const float*)d_in[30];
  const float* p2_b     = (const float*)d_in[31];
  const float* p2_g     = (const float*)d_in[32];
  const float* p2_be    = (const float*)d_in[33];

  const int Nn = in_sizes[1] / 3;        // 16384
  const int Ec = in_sizes[3] / 2;        // 262144
  const int Kn = Ec / Nn;                // 16
  const int DIN = 128, DOUT = 256, C1 = 64, C2 = 128, H1 = 32;

  // ---- bump allocator over d_ws (~394 MB total) ----
  char* wsp = (char*)d_ws;
  auto alloc = [&](size_t bytes) -> void* {
    void* p = (void*)wsp;
    wsp += (bytes + 255) & ~(size_t)255;
    return p;
  };
  _Float16* xh       = (_Float16*)alloc((size_t)Nn * DIN * 2);
  _Float16* wp_mlp1  = (_Float16*)alloc((size_t)DIN * H1 * 2);
  _Float16* wp_sc    = (_Float16*)alloc((size_t)DIN * DOUT * 2);
  _Float16* wp_mlp2  = (_Float16*)alloc((size_t)C2 * DOUT * 2);
  _Float16* wp_att1  = (_Float16*)alloc((size_t)C1 * C1 * 2);
  _Float16* wp_post1 = (_Float16*)alloc((size_t)C1 * C1 * 2);
  _Float16* wp_att2  = (_Float16*)alloc((size_t)C2 * C2 * 2);
  _Float16* wp_post2 = (_Float16*)alloc((size_t)C2 * C2 * 2);
  float*    stats    = (float*)alloc(16 * 256 * 4);      // mu/rv slots
  float*    scraw    = (float*)alloc((size_t)Nn * DOUT * 4);
  float*    m2raw    = (float*)alloc((size_t)Nn * DOUT * 4);
  float*    nraw     = (float*)alloc((size_t)Nn * DOUT * 4);   // reused raw buffer
  float*    h1F      = (float*)alloc((size_t)Nn * H1 * 4);
  float*    hl1F     = (float*)alloc((size_t)Nn * C1 * 4);
  _Float16* aggH     = (_Float16*)alloc((size_t)Nn * C2 * 2);  // reused both LFAs
  _Float16* hl2H     = (_Float16*)alloc((size_t)Nn * C2 * 2);
  float*    rel      = (float*)alloc((size_t)Ec * 8 * 4);
  float*    encraw   = (float*)alloc((size_t)Ec * C1 * 4);     // reused both LFAs
  float*    lse      = (float*)alloc((size_t)Ec * C1 * 4);     // reused both LFAs
  _Float16* localH   = (_Float16*)alloc((size_t)Ec * C2 * 2);  // reused both LFAs
  float*    att      = (float*)alloc((size_t)Ec * C2 * 4);     // reused both LFAs
  auto SL = [&](int s) { return stats + s * 256; };

  auto pack = [&](const float* W, _Float16* Wp, int Kd, int Nc) {
    int total = Kd * Nc;
    k_pack_w<<<(total + 255) / 256, 256, 0, stream>>>(W, Wp, Kd, Nc);
  };
  // NT (column tiles per wave) chosen to minimize re-reads of the HBM-resident
  // A operand: NT=8 when Nc>=128, NT=4 for Nc=64, NT=2 for Nc=32.
  auto gemm = [&](const _Float16* A, const _Float16* Wp, const float* bias, float* C,
                  int M, int Kd, int Nc) {
    int tilesN = Nc / 16;
    if (tilesN >= 8) {
      int waves = (M / 16) * (tilesN / 8);
      k_wmma_gemm<8><<<(waves + 7) / 8, 256, 0, stream>>>(A, Wp, bias, C, M, Kd, Nc);
    } else if (tilesN >= 4) {
      int waves = (M / 16) * (tilesN / 4);
      k_wmma_gemm<4><<<(waves + 7) / 8, 256, 0, stream>>>(A, Wp, bias, C, M, Kd, Nc);
    } else {
      int waves = (M / 16) * (tilesN / 2);
      k_wmma_gemm<2><<<(waves + 7) / 8, 256, 0, stream>>>(A, Wp, bias, C, M, Kd, Nc);
    }
  };
  auto colstats = [&](const float* C, int M, int Nc, float* mu, float* rv) {
    k_colstats<<<Nc, 256, 0, stream>>>(C, mu, rv, M, Nc);
  };
  auto bn = [&](const float* C, float* mu, float* rv, const float* g, const float* be,
                float* oF, _Float16* oH, int M, int Nc, int act) {
    int total = M * Nc;
    k_bn<<<(total + 255) / 256, 256, 0, stream>>>(C, mu, rv, g, be, oF, oH, total, Nc, act);
  };

  // ---- weight prep + input cast ----
  k_cast_f16<<<(Nn * DIN + 255) / 256, 256, 0, stream>>>(x, xh, Nn * DIN);
  pack(mlp1_w, wp_mlp1, DIN, H1);
  pack(sc_w,   wp_sc,   DIN, DOUT);
  pack(mlp2_w, wp_mlp2, C2,  DOUT);
  pack(att1_w, wp_att1, C1, C1);
  pack(p1_w,   wp_post1, C1, C1);
  pack(att2_w, wp_att2, C2, C2);
  pack(p2_w,   wp_post2, C2, C2);

  // ---- shortcut: x @ sc_w + b (BN deferred to final) ----
  gemm(xh, wp_sc, sc_b, scraw, Nn, DIN, DOUT);
  colstats(scraw, Nn, DOUT, SL(0), SL(1));

  // ---- mlp1: h = lrelu(BN(x @ mlp1_w + b)) -> h1F [N,32] ----
  gemm(xh, wp_mlp1, mlp1_b, nraw, Nn, DIN, H1);
  colstats(nraw, Nn, H1, SL(2), SL(3));
  bn(nraw, SL(2), SL(3), mlp1_g, mlp1_be, h1F, nullptr, Nn, H1, 1);

  // ---- darboux features (shared by both LFAs) ----
  k_darboux<<<(Ec + 255) / 256, 256, 0, stream>>>(pos, normals, ei, rel, Ec);

  // ================= LFA 1 (c = 64) =================
  k_enc<<<(Ec * (C1 / 2) + 255) / 256, 256, 0, stream>>>(rel, e1_w, e1_b, encraw, Ec, C1 / 2);
  colstats(encraw, Ec, C1 / 2, SL(4), SL(5));
  bn(encraw, SL(4), SL(5), e1_g, e1_be, lse, nullptr, Ec, C1 / 2, 1);
  k_local<<<(Ec * C1 + 255) / 256, 256, 0, stream>>>(h1F, lse, ei, localH, Ec, C1 / 2);
  gemm(localH, wp_att1, nullptr, att, Ec, C1, C1);
  k_agg<<<(Nn * C1 + 255) / 256, 256, 0, stream>>>(att, h1F, lse, ei, aggH, Nn, Kn, C1, C1 / 2);
  gemm(aggH, wp_post1, p1_b, nraw, Nn, C1, C1);
  colstats(nraw, Nn, C1, SL(6), SL(7));
  bn(nraw, SL(6), SL(7), p1_g, p1_be, hl1F, nullptr, Nn, C1, 1);

  // ================= LFA 2 (c = 128) =================
  k_enc<<<(Ec * (C2 / 2) + 255) / 256, 256, 0, stream>>>(rel, e2_w, e2_b, encraw, Ec, C2 / 2);
  colstats(encraw, Ec, C2 / 2, SL(8), SL(9));
  bn(encraw, SL(8), SL(9), e2_g, e2_be, lse, nullptr, Ec, C2 / 2, 1);
  k_local<<<(Ec * C2 + 255) / 256, 256, 0, stream>>>(hl1F, lse, ei, localH, Ec, C2 / 2);
  gemm(localH, wp_att2, nullptr, att, Ec, C2, C2);
  k_agg<<<(Nn * C2 + 255) / 256, 256, 0, stream>>>(att, hl1F, lse, ei, aggH, Nn, Kn, C2, C2 / 2);
  gemm(aggH, wp_post2, p2_b, nraw, Nn, C2, C2);
  colstats(nraw, Nn, C2, SL(10), SL(11));
  bn(nraw, SL(10), SL(11), p2_g, p2_be, nullptr, hl2H, Nn, C2, 1);

  // ---- mlp2 + residual + final LeakyReLU ----
  gemm(hl2H, wp_mlp2, mlp2_b, m2raw, Nn, C2, DOUT);
  colstats(m2raw, Nn, DOUT, SL(12), SL(13));
  k_final<<<(Nn * DOUT + 255) / 256, 256, 0, stream>>>(
      m2raw, scraw, SL(12), SL(13), mlp2_g, mlp2_be,
      SL(0), SL(1), sc_g, sc_be, (float*)d_out, Nn * DOUT, DOUT);
}
